// RoIHeads_40321152975411
// MI455X (gfx1250) — matrix operand order repaired and checked by reading, not compiled, not run on hardware
//
#include <hip/hip_runtime.h>
#include <hip/hip_bf16.h>
#include <math.h>

typedef __attribute__((ext_vector_type(16))) _Float16 v16h;
typedef __attribute__((ext_vector_type(8)))  _Float16 v8h;
typedef __attribute__((ext_vector_type(8)))  float    v8f;

#define IMGSZ    800.0f
#define NC       91
#define KFG      90
#define SCORE_TH 0.05f
#define NMS_TH   0.5f
#define NDETS    100
#define PRE_NMS  1000
#define MIN_SZ   0.01f
#define NPROP    1000
#define MPAD     1024
#define DFEAT    12544   /* 256*7*7 */
#define RHID     1024
#define FH       50
#define DW_CLIP  4.135166556742356f  /* ln(1000/16) */

// Low 32 bits of a flat shared-aperture address are the LDS byte offset
// (ISA 10.2).
static __device__ __forceinline__ uint32_t to_lds(const void* p) {
  return (uint32_t)(uintptr_t)p;
}

// CDNA5 async copy: 16B per lane, global -> LDS, tracked by ASYNCcnt.
#define ASYNC_LD_B128(ldsoff, gptr)                                   \
  asm volatile("global_load_async_to_lds_b128 %0, %1, off"            \
               :: "v"(ldsoff), "v"(gptr) : "memory")
#define WAIT_ASYNC_3() asm volatile("s_wait_asynccnt 0x3" ::: "memory")
#define WAIT_ASYNC_0() asm volatile("s_wait_asynccnt 0x0" ::: "memory")

// ---------------------------------------------------------------- RoIAlign ->
// writes A matrix [MPAD][DFEAT] as f16, rows >= NPROP zeroed for GEMM padding.
__global__ void roi_align_kernel(const float* __restrict__ feat,
                                 const float* __restrict__ prop,
                                 _Float16* __restrict__ A) {
  int n = blockIdx.x;          // 0..1023
  int c = threadIdx.x;         // 0..255 (channel)
  __shared__ int   sy0[14], sx0[14];
  __shared__ float sly[14], slx[14];
  _Float16* out = A + (size_t)n * DFEAT + c * 49;
  if (n >= NPROP) {
    for (int t = 0; t < 49; ++t) out[t] = (_Float16)0.0f;
    return;
  }
  if (threadIdx.x < 28) {
    int s = threadIdx.x % 14;
    bool isx = threadIdx.x >= 14;
    float x1 = prop[n * 4 + 0] * 0.0625f;
    float y1 = prop[n * 4 + 1] * 0.0625f;
    float x2 = prop[n * 4 + 2] * 0.0625f;
    float y2 = prop[n * 4 + 3] * 0.0625f;
    float rw = fmaxf(x2 - x1, 1.0f);
    float rh = fmaxf(y2 - y1, 1.0f);
    float t = ((float)s + 0.5f) / 14.0f;
    float p = isx ? (x1 + t * rw) : (y1 + t * rh);
    float pc = fminf(fmaxf(p, 0.0f), (float)(FH - 1));
    float p0 = fminf(fmaxf(floorf(pc), 0.0f), (float)(FH - 2));
    float l = pc - p0;
    if (isx) { sx0[s] = (int)p0; slx[s] = l; }
    else     { sy0[s] = (int)p0; sly[s] = l; }
  }
  __syncthreads();
  const float* f = feat + (size_t)c * (FH * FH);
  for (int i = 0; i < 7; ++i)
    for (int j = 0; j < 7; ++j) {
      float acc = 0.0f;
      #pragma unroll
      for (int sy = 0; sy < 2; ++sy)
        #pragma unroll
        for (int sx = 0; sx < 2; ++sx) {
          int yy = i * 2 + sy, xx = j * 2 + sx;
          int y0 = sy0[yy], x0 = sx0[xx];
          float ly = sly[yy], lx = slx[xx];
          float f00 = f[y0 * FH + x0];
          float f01 = f[y0 * FH + x0 + 1];
          float f10 = f[(y0 + 1) * FH + x0];
          float f11 = f[(y0 + 1) * FH + x0 + 1];
          acc += f00 * (1.f - ly) * (1.f - lx) + f01 * (1.f - ly) * lx +
                 f10 * ly * (1.f - lx) + f11 * ly * lx;
        }
      out[i * 7 + j] = (_Float16)(acc * 0.25f);
    }
}

// ------------------------------------------- transpose f32 [K][N] -> f16 [N][K]
__global__ void transpose_f32_to_f16(const float* __restrict__ Wsrc,
                                     _Float16* __restrict__ BT, int K, int N) {
  __shared__ float tile[32][33];
  int k0 = blockIdx.x * 32;
  int n0 = blockIdx.y * 32;
  int tx = threadIdx.x, ty = threadIdx.y;  // 32 x 8
  #pragma unroll
  for (int r = 0; r < 32; r += 8)
    tile[ty + r][tx] = Wsrc[(size_t)(k0 + ty + r) * N + (n0 + tx)];
  __syncthreads();
  #pragma unroll
  for (int r = 0; r < 32; r += 8)
    BT[(size_t)(n0 + ty + r) * K + (k0 + tx)] = (_Float16)tile[tx][ty + r];
}

// ----------------------------------------------------- LDS-staged WMMA GEMM
// A: [M][K] f16 row-major, BT: [N][K] f16 (B transposed).
// Block = 8 waves -> 128x64 macro-tile; wave -> 64x16 (4 chained WMMA tiles).
// K-slices staged via async-to-LDS, 2-deep software pipeline with static
// buffer roles (no per-iteration buffer selects / branches).
__device__ __forceinline__ v16h pack16(v8h lo, v8h hi) {
  v16h r;
  #pragma unroll
  for (int i = 0; i < 8; ++i) { r[i] = lo[i]; r[i + 8] = hi[i]; }
  return r;
}

#define LDK 40   /* 32 + 8 pad; 80B row keeps 16B chunk alignment */

__global__ void __launch_bounds__(256)
gemm_wmma_f16(const _Float16* __restrict__ A,
              const _Float16* __restrict__ BT,
              const float* __restrict__ bias,
              _Float16* __restrict__ O16,
              float* __restrict__ O32,
              int Ntot, int K) {
  __shared__ _Float16 sA[2][128 * LDK];
  __shared__ _Float16 sB[2][64 * LDK];

  int tid = threadIdx.x;
  int lane = tid & 31, wid = tid >> 5;
  int nb = Ntot >> 6;                    // # of 64-col macro tiles
  int bm = blockIdx.x / nb, bn = blockIdx.x % nb;
  int row0 = bm * 128, col0 = bn * 64;

  int wmL = wid >> 2, wnL = wid & 3;     // 2 x 4 wave grid
  int m16 = lane & 15, kh = lane >> 4;

  // Loader assignment: 16B chunks. A slice 128x32 -> 512 chunks (2/thread),
  // B slice 64x32 -> 256 chunks (1/thread).
  int rA0 = tid >> 2;            // 0..63
  int rA1 = rA0 + 64;            // 64..127
  int kcA = (tid & 3) * 8;

  const _Float16* gA0 = A + (size_t)(row0 + rA0) * K + kcA;
  const _Float16* gA1 = A + (size_t)(row0 + rA1) * K + kcA;
  const _Float16* gB  = BT + (size_t)(col0 + rA0) * K + kcA;
  uint32_t lA0[2], lA1[2], lB[2];
  const _Float16* abuf[2];
  const _Float16* bbuf[2];
  #pragma unroll
  for (int b = 0; b < 2; ++b) {
    lA0[b] = to_lds(&sA[b][rA0 * LDK + kcA]);
    lA1[b] = to_lds(&sA[b][rA1 * LDK + kcA]);
    lB[b]  = to_lds(&sB[b][rA0 * LDK + kcA]);
    abuf[b] = &sA[b][(wmL * 64 + m16) * LDK + kh * 8];
    bbuf[b] = &sB[b][(wnL * 16 + m16) * LDK + kh * 8];
  }

  v8f acc[4];
  #pragma unroll
  for (int t = 0; t < 4; ++t) acc[t] = (v8f){};

  auto issue = [&](int b) {
    ASYNC_LD_B128(lA0[b], gA0);
    ASYNC_LD_B128(lA1[b], gA1);
    ASYNC_LD_B128(lB[b], gB);
    gA0 += 32; gA1 += 32; gB += 32;
  };
  auto compute = [&](int b) {
    v16h bfrag = pack16(*(const v8h*)(bbuf[b]), *(const v8h*)(bbuf[b] + 16));
    #pragma unroll
    for (int t = 0; t < 4; ++t) {
      v16h af = pack16(*(const v8h*)(abuf[b] + t * 16 * LDK),
                       *(const v8h*)(abuf[b] + t * 16 * LDK + 16));
      acc[t] = __builtin_amdgcn_wmma_f32_16x16x32_f16(
          false, af, false, bfrag, (short)0, acc[t], false, false);
    }
  };

  int nsteps = K >> 5;   // even (K = 12544 or 1024), >= 2
  issue(0);              // slice 0 -> buf0
  issue(1);              // slice 1 -> buf1   (6 async ops outstanding)

  for (int s = 0; s + 2 < nsteps; s += 2) {
    WAIT_ASYNC_3();      // buf0 slice ready (3 newest still in flight)
    __syncthreads();
    compute(0);
    __syncthreads();     // everyone done reading buf0
    issue(0);            // slice s+2 -> buf0
    WAIT_ASYNC_3();      // buf1 slice ready
    __syncthreads();
    compute(1);
    __syncthreads();     // everyone done reading buf1
    issue(1);            // slice s+3 -> buf1
  }
  WAIT_ASYNC_3();
  __syncthreads();
  compute(0);
  WAIT_ASYNC_0();
  __syncthreads();
  compute(1);

  int col = col0 + wnL * 16 + m16;
  float bv = bias[col];
  #pragma unroll
  for (int t = 0; t < 4; ++t) {
    int rbase = row0 + wmL * 64 + t * 16 + kh * 8;
    #pragma unroll
    for (int v = 0; v < 8; ++v) {
      float x = fmaxf(acc[t][v] + bv, 0.0f);
      size_t o = (size_t)(rbase + v) * Ntot + col;
      if (O16) O16[o] = (_Float16)x;
      if (O32) O32[o] = x;
    }
  }
}

// --------------------------------------------------------------- cls/bbox heads
__global__ void heads_kernel(const float* __restrict__ X,   // [MPAD][RHID] f32
                             const float* __restrict__ Wc, const float* __restrict__ bc,
                             const float* __restrict__ Wb, const float* __restrict__ bb,
                             float* __restrict__ logits, float* __restrict__ deltas) {
  __shared__ float xs[RHID];
  int n = blockIdx.x;
  for (int k = threadIdx.x; k < RHID; k += blockDim.x)
    xs[k] = X[(size_t)n * RHID + k];
  __syncthreads();
  int j = threadIdx.x;  // 512 threads, 455 active
  if (j < NC) {
    float s = bc[j];
    for (int k = 0; k < RHID; ++k) s = fmaf(xs[k], Wc[(size_t)k * NC + j], s);
    logits[(size_t)n * NC + j] = s;
  } else if (j < NC + 4 * NC) {
    int jj = j - NC;
    float s = bb[jj];
    for (int k = 0; k < RHID; ++k) s = fmaf(xs[k], Wb[(size_t)k * (4 * NC) + jj], s);
    deltas[(size_t)n * 4 * NC + jj] = s;
  }
}

// ------------------------------------------------------------------- softmax
__global__ void softmax_kernel(const float* __restrict__ logits,
                               float* __restrict__ scores) {
  int n = blockIdx.x * blockDim.x + threadIdx.x;
  if (n >= NPROP) return;
  const float* l = logits + (size_t)n * NC;
  float m = l[0];
  for (int k = 1; k < NC; ++k) m = fmaxf(m, l[k]);
  float s = 0.0f;
  for (int k = 0; k < NC; ++k) s += expf(l[k] - m);
  float inv = 1.0f / s;
  for (int k = 0; k < NC; ++k) scores[(size_t)n * NC + k] = expf(l[k] - m) * inv;
}

// ------------------------------------------- decode boxes + build candidates
__global__ void decode_kernel(const float* __restrict__ deltas,
                              const float* __restrict__ scores,
                              const float* __restrict__ prop,
                              float* __restrict__ cand_sm,
                              float* __restrict__ cand_box) {
  int e = blockIdx.x * blockDim.x + threadIdx.x;
  if (e >= NPROP * NC) return;
  int n = e / NC, k = e % NC;
  if (k == 0) return;  // background class dropped
  float px1 = prop[n * 4 + 0], py1 = prop[n * 4 + 1];
  float px2 = prop[n * 4 + 2], py2 = prop[n * 4 + 3];
  float w = px2 - px1, h = py2 - py1;
  float cx = px1 + 0.5f * w, cy = py1 + 0.5f * h;
  const float* d = deltas + (size_t)n * 4 * NC + k * 4;
  float dx = d[0] * 0.1f, dy = d[1] * 0.1f;
  float dw = fminf(d[2] * 0.2f, DW_CLIP);
  float dh = fminf(d[3] * 0.2f, DW_CLIP);
  float pcx = dx * w + cx, pcy = dy * h + cy;
  float pw = expf(dw) * w, ph = expf(dh) * h;
  float x1 = fminf(fmaxf(pcx - 0.5f * pw, 0.0f), IMGSZ);
  float y1 = fminf(fmaxf(pcy - 0.5f * ph, 0.0f), IMGSZ);
  float x2 = fminf(fmaxf(pcx + 0.5f * pw, 0.0f), IMGSZ);
  float y2 = fminf(fmaxf(pcy + 0.5f * ph, 0.0f), IMGSZ);
  int ci = n * KFG + (k - 1);
  float sf = scores[(size_t)n * NC + k];
  bool valid = (sf > SCORE_TH) && ((x2 - x1) > MIN_SZ) && ((y2 - y1) > MIN_SZ);
  cand_sm[ci] = valid ? sf : -1.0f;
  cand_box[ci * 4 + 0] = x1;
  cand_box[ci * 4 + 1] = y1;
  cand_box[ci * 4 + 2] = x2;
  cand_box[ci * 4 + 3] = y2;
}

// ------------------- exact lax.top_k(PRE_NMS): value desc, ties -> lowest index
__global__ void topk1000_kernel(float* __restrict__ sm,
                                int* __restrict__ sel_idx,
                                float* __restrict__ sel_val) {
  __shared__ float sv[1024];
  __shared__ int   si[1024];
  int t = threadIdx.x;
  for (int kk = 0; kk < PRE_NMS; ++kk) {
    float bv = -1e30f; int bi = 0x7fffffff;
    for (int i = t; i < NPROP * KFG; i += 1024) {
      float v = sm[i];
      if (v > bv || (v == bv && i < bi)) { bv = v; bi = i; }
    }
    sv[t] = bv; si[t] = bi;
    __syncthreads();
    for (int s = 512; s > 0; s >>= 1) {
      if (t < s) {
        float ov = sv[t + s]; int oi = si[t + s];
        if (ov > sv[t] || (ov == sv[t] && oi < si[t])) { sv[t] = ov; si[t] = oi; }
      }
      __syncthreads();
    }
    if (t == 0) {
      sel_idx[kk] = si[0];
      sel_val[kk] = sv[0];
      sm[si[0]] = -1e30f;   // remove
    }
    __syncthreads();
  }
}

// ----------------------- NMS (sequential scan, class-offset boxes) + final top-100
__global__ void nms_final_kernel(const int* __restrict__ sel_idx,
                                 const float* __restrict__ sel_val,
                                 const float* __restrict__ cand_box,
                                 float* __restrict__ out) {
  __shared__ float bx[PRE_NMS][4];
  __shared__ float area[PRE_NMS];
  __shared__ int   keep[PRE_NMS];
  __shared__ float mv[PRE_NMS];
  __shared__ float rv[1024];
  __shared__ int   ri[1024];
  int t = threadIdx.x;
  for (int j = t; j < PRE_NMS; j += 1024) {
    int ci = sel_idx[j];
    int label = ci % KFG + 1;
    float off = (float)label * (IMGSZ + 1.0f);
    float x1 = cand_box[ci * 4 + 0] + off;
    float y1 = cand_box[ci * 4 + 1] + off;
    float x2 = cand_box[ci * 4 + 2] + off;
    float y2 = cand_box[ci * 4 + 3] + off;
    bx[j][0] = x1; bx[j][1] = y1; bx[j][2] = x2; bx[j][3] = y2;
    area[j] = (x2 - x1) * (y2 - y1);
    keep[j] = 1;
  }
  __syncthreads();
  for (int i = 0; i < PRE_NMS; ++i) {
    int ki = keep[i];
    float ix1 = bx[i][0], iy1 = bx[i][1], ix2 = bx[i][2], iy2 = bx[i][3];
    float ia = area[i];
    if (ki && t > i && t < PRE_NMS && keep[t]) {
      float xx1 = fmaxf(ix1, bx[t][0]), yy1 = fmaxf(iy1, bx[t][1]);
      float xx2 = fminf(ix2, bx[t][2]), yy2 = fminf(iy2, bx[t][3]);
      float iw = fmaxf(xx2 - xx1, 0.0f), ih = fmaxf(yy2 - yy1, 0.0f);
      float inter = iw * ih;
      float iou = inter / (ia + area[t] - inter + 1e-9f);
      if (iou > NMS_TH) keep[t] = 0;
    }
    __syncthreads();
  }
  for (int j = t; j < PRE_NMS; j += 1024) mv[j] = keep[j] ? sel_val[j] : -1.0f;
  __syncthreads();
  for (int kk = 0; kk < NDETS; ++kk) {
    float bv; int bi;
    if (t < PRE_NMS) { bv = mv[t]; bi = t; } else { bv = -1e30f; bi = 0x7fffffff; }
    rv[t] = bv; ri[t] = bi;
    __syncthreads();
    for (int s = 512; s > 0; s >>= 1) {
      if (t < s) {
        if (rv[t + s] > rv[t] || (rv[t + s] == rv[t] && ri[t + s] < ri[t])) {
          rv[t] = rv[t + s]; ri[t] = ri[t + s];
        }
      }
      __syncthreads();
    }
    if (t == 0) {
      int j = ri[0];
      int ci = sel_idx[j];
      int label = ci % KFG + 1;
      out[kk * 5 + 0] = cand_box[ci * 4 + 0];
      out[kk * 5 + 1] = cand_box[ci * 4 + 1];
      out[kk * 5 + 2] = cand_box[ci * 4 + 2];
      out[kk * 5 + 3] = cand_box[ci * 4 + 3];
      out[kk * 5 + 4] = rv[0];
      out[NDETS * 5 + kk] = (float)label;
      mv[j] = -1e30f;
    }
    __syncthreads();
  }
}

// ==================================================================== launch
extern "C" void kernel_launch(void* const* d_in, const int* in_sizes, int n_in,
                              void* d_out, int out_size, void* d_ws, size_t ws_size,
                              hipStream_t stream) {
  const float* features = (const float*)d_in[0];
  const float* proposals = (const float*)d_in[1];
  const float* w_fc1 = (const float*)d_in[2];
  const float* b_fc1 = (const float*)d_in[3];
  const float* w_fc2 = (const float*)d_in[4];
  const float* b_fc2 = (const float*)d_in[5];
  const float* w_cls = (const float*)d_in[6];
  const float* b_cls = (const float*)d_in[7];
  const float* w_bbox = (const float*)d_in[8];
  const float* b_bbox = (const float*)d_in[9];

  char* W = (char*)d_ws;
  size_t off = 0;
  auto take = [&](size_t bytes) {
    size_t cur = off;
    off += (bytes + 255) & ~(size_t)255;
    return cur;
  };
  _Float16* A16  = (_Float16*)(W + take((size_t)MPAD * DFEAT * 2));
  _Float16* BT1  = (_Float16*)(W + take((size_t)RHID * DFEAT * 2));
  _Float16* X1   = (_Float16*)(W + take((size_t)MPAD * RHID * 2));
  _Float16* BT2  = (_Float16*)(W + take((size_t)RHID * RHID * 2));
  float* X2      = (float*)(W + take((size_t)MPAD * RHID * 4));
  float* logits  = (float*)(W + take((size_t)NPROP * NC * 4));
  float* scores  = (float*)(W + take((size_t)NPROP * NC * 4));
  float* deltas  = (float*)(W + take((size_t)NPROP * NC * 16));
  float* cand_sm = (float*)(W + take((size_t)NPROP * KFG * 4));
  float* cand_box= (float*)(W + take((size_t)NPROP * KFG * 16));
  int*   sel_idx = (int*)(W + take((size_t)PRE_NMS * 4));
  float* sel_val = (float*)(W + take((size_t)PRE_NMS * 4));
  (void)ws_size; (void)in_sizes; (void)n_in; (void)out_size;

  // 1. RoIAlign -> f16 A matrix (rows padded to 1024)
  roi_align_kernel<<<MPAD, 256, 0, stream>>>(features, proposals, A16);

  // 2. Weight transpose+convert to f16 [N][K]
  transpose_f32_to_f16<<<dim3(DFEAT / 32, RHID / 32), dim3(32, 8), 0, stream>>>(
      w_fc1, BT1, DFEAT, RHID);
  transpose_f32_to_f16<<<dim3(RHID / 32, RHID / 32), dim3(32, 8), 0, stream>>>(
      w_fc2, BT2, RHID, RHID);

  // 3. fc1: (1024x12544) x (12544x1024), bias+ReLU -> f16
  //    grid = (1024/128 row blocks) x (1024/64 col blocks) = 128 blocks
  gemm_wmma_f16<<<128, 256, 0, stream>>>(A16, BT1, b_fc1, X1, nullptr, RHID, DFEAT);

  // 4. fc2: (1024x1024) x (1024x1024), bias+ReLU -> f32
  gemm_wmma_f16<<<128, 256, 0, stream>>>(X1, BT2, b_fc2, nullptr, X2, RHID, RHID);

  // 5. cls/bbox heads in f32
  heads_kernel<<<NPROP, 512, 0, stream>>>(X2, w_cls, b_cls, w_bbox, b_bbox,
                                          logits, deltas);

  // 6. softmax
  softmax_kernel<<<(NPROP + 127) / 128, 128, 0, stream>>>(logits, scores);

  // 7. decode + candidate scoring
  decode_kernel<<<(NPROP * NC + 255) / 256, 256, 0, stream>>>(
      deltas, scores, proposals, cand_sm, cand_box);

  // 8. top-1000 (exact lax.top_k semantics)
  topk1000_kernel<<<1, 1024, 0, stream>>>(cand_sm, sel_idx, sel_val);

  // 9. NMS scan + final top-100 + output write
  nms_final_kernel<<<1, 1024, 0, stream>>>(sel_idx, sel_val, cand_box,
                                           (float*)d_out);
}
